// KVCacheAttention_43473658970633
// MI455X (gfx1250) — matrix operand order repaired and checked by reading, not compiled, not run on hardware
//
#include <hip/hip_runtime.h>
#include <hip/hip_bf16.h>

// ---------------------------------------------------------------------------
// KV-cache attention for MI455X (gfx1250, wave32, WMMA).
// B=4, Sq=2048, Sc=6144, Skv=8192, D=1024. All matmuls in bf16 WMMA
// (v_wmma_f32_16x16x32_bf16), fp32 accumulation, flash-attention online
// softmax so the 256MB score tensor is never materialized.
// L2 (192MB) holds bf16 K+V for all batches (134MB), so q-tiles of 16 rows
// re-streaming K/V stay on-chip. global_prefetch_b8 hides first-touch latency.
// ---------------------------------------------------------------------------

typedef __bf16 bf16;
typedef __attribute__((ext_vector_type(16))) __bf16 v16bf;
typedef __attribute__((ext_vector_type(8)))  __bf16 v8bf;
typedef __attribute__((ext_vector_type(8)))  float  v8f;

#define D_    1024
#define SQ_   2048
#define SC_   6144
#define SKV_  8192
#define B_    4

static __device__ __forceinline__ v16bf cat8(v8bf lo, v8bf hi) {
  return __builtin_shufflevector(lo, hi, 0,1,2,3,4,5,6,7,8,9,10,11,12,13,14,15);
}
static __device__ __forceinline__ v8f wmma_bf16(v16bf a, v16bf b, v8f c) {
  // D = A(16x32) * B(32x16) + C, fp32 accumulate
  return __builtin_amdgcn_wmma_f32_16x16x32_bf16(false, a, false, b, (short)0, c, false, false);
}

// 16-lane row reductions (rows are 16 consecutive lanes of a wave32)
static __device__ __forceinline__ float rmax16(float v) {
  #pragma unroll
  for (int m = 8; m >= 1; m >>= 1) v = fmaxf(v, __shfl_xor(v, m, 32));
  return v;
}
static __device__ __forceinline__ float rsum16(float v) {
  #pragma unroll
  for (int m = 8; m >= 1; m >>= 1) v += __shfl_xor(v, m, 32);
  return v;
}

// ---------------------------------------------------------------------------
// Elementwise f32 -> bf16
__global__ void cvt_f32_bf16(const float* __restrict__ s, bf16* __restrict__ d, int n) {
  int i = blockIdx.x * 256 + threadIdx.x;
  if (i < n) d[i] = (bf16)s[i];
}

// cached_key [B][SC][D] f32 -> K bf16 [B][SKV][D] rows [0,SC)
__global__ void cvt_cache_k(const float* __restrict__ s, bf16* __restrict__ d) {
  int i = blockIdx.x * 256 + threadIdx.x;           // over B*SC*D (divisible by 256)
  int b = i / (SC_ * D_);
  int rem = i - b * (SC_ * D_);
  d[(size_t)b * SKV_ * D_ + rem] = (bf16)s[i];
}

// cached_value [B][SC][D] f32 -> Vt bf16 [B][D][SKV] kv in [0,SC) (LDS-tiled transpose)
__global__ void cvt_transpose_v(const float* __restrict__ s, bf16* __restrict__ d) {
  __shared__ bf16 tile[32][33];
  int b  = blockIdx.z;
  int d0 = blockIdx.x * 32, s0 = blockIdx.y * 32;
  int tx = threadIdx.x, ty = threadIdx.y;           // 32 x 8
  #pragma unroll
  for (int k = 0; k < 4; ++k) {
    int srow = s0 + ty + k * 8;
    tile[ty + k * 8][tx] = (bf16)s[((size_t)b * SC_ + srow) * D_ + d0 + tx];
  }
  __syncthreads();
  #pragma unroll
  for (int k = 0; k < 4; ++k) {
    int dr = d0 + ty + k * 8;
    d[(size_t)b * D_ * SKV_ + (size_t)dr * SKV_ + s0 + tx] = tile[tx][ty + k * 8];
  }
}

// ---------------------------------------------------------------------------
// Projection GEMM: out = X @ W^T + bias.  X:[B*SQ][D] bf16, W:[D][D] bf16 (row
// n holds weights for output feature n, contraction contiguous).
// Block tile 128x128, 8 waves, each wave 64(M)x32(N) = 4x2 WMMA tiles.
// mode 0: Q[b][s][n]   mode 1: K[b][SC+s][n]   mode 2: Vt[b][n][SC+s]
__global__ __launch_bounds__(256) void proj_gemm(const bf16* __restrict__ X,
                                                 const bf16* __restrict__ W,
                                                 const float* __restrict__ bias,
                                                 bf16* __restrict__ out, int mode) {
  int lane = threadIdx.x & 31, wid = threadIdx.x >> 5;
  int half = lane >> 4, lm = lane & 15;
  int blockM = blockIdx.y * 128, blockN = blockIdx.x * 128;
  int m_off = (wid >> 2) * 64, n_off = (wid & 3) * 32;

  v8f c[4][2];
  #pragma unroll
  for (int i = 0; i < 4; ++i)
    #pragma unroll
    for (int j = 0; j < 2; ++j) c[i][j] = (v8f)0.0f;

  for (int kk = 0; kk < D_; kk += 32) {
    v16bf a[4], bm[2];
    #pragma unroll
    for (int i = 0; i < 4; ++i) {
      const bf16* p = X + (size_t)(blockM + m_off + i * 16 + lm) * D_ + kk + 16 * half;
      v8bf lo = *(const v8bf*)p;
      v8bf hi = *(const v8bf*)(p + 8);
      a[i] = cat8(lo, hi);
      __builtin_prefetch(p + 128, 0, 0);   // 4 k-steps ahead, global_prefetch_b8
    }
    #pragma unroll
    for (int j = 0; j < 2; ++j) {
      const bf16* q = W + (size_t)(blockN + n_off + j * 16 + lm) * D_ + kk + 16 * half;
      bm[j] = *(const v16bf*)q;
      __builtin_prefetch(q + 128, 0, 0);
    }
    #pragma unroll
    for (int i = 0; i < 4; ++i)
      #pragma unroll
      for (int j = 0; j < 2; ++j) c[i][j] = wmma_bf16(a[i], bm[j], c[i][j]);
  }

  #pragma unroll
  for (int i = 0; i < 4; ++i) {
    #pragma unroll
    for (int j = 0; j < 2; ++j) {
      int n = blockN + n_off + j * 16 + lm;
      float bv = bias[n];
      #pragma unroll
      for (int v = 0; v < 8; ++v) {
        int r = blockM + m_off + i * 16 + v + 8 * half;  // global M row
        int b = r >> 11, s = r & 2047;                   // SQ = 2048
        bf16 o = (bf16)(c[i][j][v] + bv);
        if (mode == 0)      out[((size_t)b * SQ_ + s) * D_ + n] = o;
        else if (mode == 1) out[((size_t)b * SKV_ + SC_ + s) * D_ + n] = o;
        else                out[(size_t)b * D_ * SKV_ + (size_t)n * SKV_ + SC_ + s] = o;
      }
    }
  }
}

// ---------------------------------------------------------------------------
// Flash attention. One block = (batch b, 16 query rows). 8 waves; wave w owns
// output d-slice [w*128, w*128+128). Score contraction (D=1024) is split
// across waves (128 each) and reduced in LDS; online softmax; P@V per wave.
__global__ __launch_bounds__(256) void flash_attn(const bf16* __restrict__ Q,
                                                  const bf16* __restrict__ K,
                                                  const bf16* __restrict__ Vt,
                                                  float* __restrict__ out) {
  const float SCALE = 0.125f;               // 1/sqrt(HEAD_DIM=64)
  const float L2E   = 1.44269504088896f;

  __shared__ float Spart[8][16][64];
  __shared__ __align__(32) bf16 Pt[16][64];
  __shared__ float rowM[16], rowL[16], rowScale[16];

  int b  = blockIdx.y;
  int q0 = blockIdx.x * 16;
  int tid = threadIdx.x, lane = tid & 31, wid = tid >> 5;
  int half = lane >> 4, lm = lane & 15;

  if (tid < 16) { rowM[tid] = -__builtin_inff(); rowL[tid] = 0.0f; }

  v8f acc[8];
  #pragma unroll
  for (int t = 0; t < 8; ++t) acc[t] = (v8f)0.0f;

  // Q A-fragments for this wave's d-slice, reused every kv iteration.
  v16bf qa[4];
  #pragma unroll
  for (int ks = 0; ks < 4; ++ks) {
    const bf16* p = Q + ((size_t)b * SQ_ + q0 + lm) * D_ + wid * 128 + ks * 32 + 16 * half;
    v8bf lo = *(const v8bf*)p;
    v8bf hi = *(const v8bf*)(p + 8);
    qa[ks] = cat8(lo, hi);
  }
  __syncthreads();

  for (int kv0 = 0; kv0 < SKV_; kv0 += 64) {
    // ---- partial scores over this wave's 128-wide d-slice: 16 WMMAs
    v8f sc[4];
    #pragma unroll
    for (int nt = 0; nt < 4; ++nt) sc[nt] = (v8f)0.0f;
    #pragma unroll
    for (int nt = 0; nt < 4; ++nt) {
      const bf16* kp = K + ((size_t)b * SKV_ + kv0 + nt * 16 + lm) * D_ + wid * 128 + 16 * half;
      __builtin_prefetch(kp + (size_t)64 * D_, 0, 0);  // next kv tile (speculative)
      #pragma unroll
      for (int ks = 0; ks < 4; ++ks) {
        v16bf kb = *(const v16bf*)(kp + ks * 32);
        sc[nt] = wmma_bf16(qa[ks], kb, sc[nt]);
      }
    }
    #pragma unroll
    for (int nt = 0; nt < 4; ++nt)
      #pragma unroll
      for (int v = 0; v < 8; ++v)
        Spart[wid][v + 8 * half][nt * 16 + lm] = sc[nt][v];
    __syncthreads();

    // ---- reduce partials + online softmax (row = 16 consecutive lanes)
    {
      int r = tid >> 4, c0 = tid & 15;
      float s4[4], p4[4];
      float mloc = -__builtin_inff();
      #pragma unroll
      for (int j = 0; j < 4; ++j) {
        int col = c0 + j * 16;
        float s = 0.0f;
        #pragma unroll
        for (int wv = 0; wv < 8; ++wv) s += Spart[wv][r][col];
        s *= SCALE;
        s4[j] = s;
        mloc = fmaxf(mloc, s);
      }
      float mt = rmax16(mloc);
      float mo = rowM[r];
      float mn = fmaxf(mo, mt);
      float sl = 0.0f;
      #pragma unroll
      for (int j = 0; j < 4; ++j) { p4[j] = exp2f((s4[j] - mn) * L2E); sl += p4[j]; }
      float st  = rsum16(sl);
      float scl = exp2f((mo - mn) * L2E);
      if (c0 == 0) {
        rowM[r] = mn;
        rowL[r] = rowL[r] * scl + st;
        rowScale[r] = scl;
      }
      #pragma unroll
      for (int j = 0; j < 4; ++j) Pt[r][c0 + j * 16] = (bf16)p4[j];
    }
    __syncthreads();

    // ---- rescale accumulators by per-row exp(m_old - m_new)
    float scv[8];
    #pragma unroll
    for (int v = 0; v < 8; ++v) scv[v] = rowScale[v + 8 * half];
    #pragma unroll
    for (int t = 0; t < 8; ++t)
      #pragma unroll
      for (int v = 0; v < 8; ++v) acc[t][v] *= scv[v];

    // ---- P @ V for this wave's 128 columns: 16 WMMAs
    v16bf pa[2];
    #pragma unroll
    for (int ka = 0; ka < 2; ++ka) {
      int base = ka * 32 + 16 * half;
      v8bf lo = *(const v8bf*)&Pt[lm][base];
      v8bf hi = *(const v8bf*)&Pt[lm][base + 8];
      pa[ka] = cat8(lo, hi);
    }
    #pragma unroll
    for (int t = 0; t < 8; ++t) {
      const bf16* vp = Vt + (size_t)b * D_ * SKV_ +
                       (size_t)(wid * 128 + t * 16 + lm) * SKV_ + kv0 + 16 * half;
      __builtin_prefetch(vp + 64, 0, 0);               // next kv tile of Vt row
      #pragma unroll
      for (int ka = 0; ka < 2; ++ka) {
        v16bf vb = *(const v16bf*)(vp + ka * 32);
        acc[t] = wmma_bf16(pa[ka], vb, acc[t]);
      }
    }
    __syncthreads();
  }

  // ---- epilogue: divide by softmax denominator, store fp32
  float invl[8];
  #pragma unroll
  for (int v = 0; v < 8; ++v) invl[v] = 1.0f / rowL[v + 8 * half];
  #pragma unroll
  for (int t = 0; t < 8; ++t) {
    #pragma unroll
    for (int v = 0; v < 8; ++v) {
      int row = q0 + v + 8 * half;
      int col = wid * 128 + t * 16 + lm;
      out[((size_t)b * SQ_ + row) * D_ + col] = acc[t][v] * invl[v];
    }
  }
}

// ---------------------------------------------------------------------------
extern "C" void kernel_launch(void* const* d_in, const int* in_sizes, int n_in,
                              void* d_out, int out_size, void* d_ws, size_t ws_size,
                              hipStream_t stream) {
  const float* hidden = (const float*)d_in[0];
  const float* ck     = (const float*)d_in[1];
  const float* cv     = (const float*)d_in[2];
  const float* Wq     = (const float*)d_in[3];
  const float* bq     = (const float*)d_in[4];
  const float* Wk     = (const float*)d_in[5];
  const float* bk     = (const float*)d_in[6];
  const float* Wv     = (const float*)d_in[7];
  const float* bv     = (const float*)d_in[8];
  float* out = (float*)d_out;

  // Workspace layout (bytes), ~166 MB total.
  char* ws = (char*)d_ws;
  const size_t sz_h  = (size_t)B_ * SQ_ * D_ * 2;   // 16.8 MB
  const size_t sz_w  = (size_t)D_ * D_ * 2;         //  2.1 MB each
  const size_t sz_q  = sz_h;
  const size_t sz_kv = (size_t)B_ * SKV_ * D_ * 2;  // 67.1 MB each
  bf16* h_bf  = (bf16*)(ws);
  bf16* wq_bf = (bf16*)(ws + sz_h);
  bf16* wk_bf = (bf16*)(ws + sz_h + sz_w);
  bf16* wv_bf = (bf16*)(ws + sz_h + 2 * sz_w);
  bf16* q_bf  = (bf16*)(ws + sz_h + 3 * sz_w);
  bf16* k_bf  = (bf16*)(ws + sz_h + 3 * sz_w + sz_q);
  bf16* v_bf  = (bf16*)(ws + sz_h + 3 * sz_w + sz_q + sz_kv);  // Vt [B][D][SKV]

  // 1) downcast inputs
  int nh = B_ * SQ_ * D_;
  cvt_f32_bf16<<<nh / 256, 256, 0, stream>>>(hidden, h_bf, nh);
  int nw = D_ * D_;
  cvt_f32_bf16<<<nw / 256, 256, 0, stream>>>(Wq, wq_bf, nw);
  cvt_f32_bf16<<<nw / 256, 256, 0, stream>>>(Wk, wk_bf, nw);
  cvt_f32_bf16<<<nw / 256, 256, 0, stream>>>(Wv, wv_bf, nw);
  cvt_cache_k<<<(B_ * SC_ * D_) / 256, 256, 0, stream>>>(ck, k_bf);
  dim3 tgrid(D_ / 32, SC_ / 32, B_);
  cvt_transpose_v<<<tgrid, dim3(32, 8), 0, stream>>>(cv, v_bf);

  // 2) QKV projections (K/V appended into kv caches at row 6144)
  dim3 pgrid(D_ / 128, (B_ * SQ_) / 128);  // (8, 64)
  proj_gemm<<<pgrid, 256, 0, stream>>>(h_bf, wq_bf, bq, q_bf, 0);
  proj_gemm<<<pgrid, 256, 0, stream>>>(h_bf, wk_bf, bk, k_bf, 1);
  proj_gemm<<<pgrid, 256, 0, stream>>>(h_bf, wv_bf, bv, v_bf, 2);

  // 3) flash attention
  dim3 agrid(SQ_ / 16, B_);
  flash_attn<<<agrid, 256, 0, stream>>>(q_bf, k_bf, v_bf, out);
}
